// HybridDiffusionBlock_65335042506823
// MI455X (gfx1250) — matrix operand order repaired and checked by reading, not compiled, not run on hardware
//
#include <hip/hip_runtime.h>

#define N_  2048
#define A_  20
#define C_  256
#define P_  32
#define K_  32
#define H_  8
#define DH_ 32
#define NBINS_ 16

typedef __attribute__((ext_vector_type(16))) _Float16 v16h;
typedef __attribute__((ext_vector_type(8)))  _Float16 v8h;
typedef __attribute__((ext_vector_type(8)))  float    v8f;

__device__ __forceinline__ float sigmoidf_(float x) { return 1.0f / (1.0f + expf(-x)); }
__device__ __forceinline__ float geluf_(float x) {
    // jax.nn.gelu default (tanh approximation)
    float x3 = x * x * x;
    return 0.5f * x * (1.0f + tanhf(0.7978845608028654f * (x + 0.044715f * x3)));
}

// A fragment (16x32 f16, ISA 7.12.2): lane&15 -> row; halves hold
// k = half*8 + {0..7} and half*8 + 16 + {0..7}. Two 16B loads per K-step.
__device__ __forceinline__ v16h load_a_(const _Float16* __restrict__ ab, int half) {
    v8h lo = *(const v8h*)(ab + half * 8);
    v8h hi = *(const v8h*)(ab + half * 8 + 16);
    v16h af;
#pragma unroll
    for (int e = 0; e < 8; ++e) { af[e] = lo[e]; af[8 + e] = hi[e]; }
    return af;
}

// ---------------------------------------------------------------------------
// WMMA GEMM (f16 operands, f32 accumulate):
//   Out[M x F] = (addend? addend : 0) + A[M x Kd] @ W[Kd x F] + bias
// Contract (enforced by operand padding at conversion time):
//   Kd % 32 == 0, lda % 32 == 0, F % 32 == 0.  Fs <= F limits stores/bias to
//   the real column count (padded cols are never written or read).
// block = 128 (4 waves); each wave computes a 16(M) x 32(F) tile (2 acc
// sharing one A fragment); block tile = 16 x 128. Single branch-free loop:
// 2 x b128 A loads + 32 scalar f16 B loads + 2 x v_wmma per K-step.
// ---------------------------------------------------------------------------
__global__ void gemm_wmma_kernel(const _Float16* __restrict__ A, int lda,
                                 const _Float16* __restrict__ W,
                                 const float* __restrict__ bias, const float* __restrict__ addend,
                                 float* __restrict__ Out, int M, int Kd, int F, int Fs) {
    int wave = threadIdx.x >> 5;
    int lane = threadIdx.x & 31;
    int mtile = blockIdx.x * 16;
    int f0 = blockIdx.y * 128 + wave * 32;
    if (f0 >= F) return;                      // wave-uniform; no barriers in kernel
    int f1 = f0 + 16;
    int half = lane >> 4;
    int r    = lane & 15;

    int row = mtile + r;
    if (row >= M) row = M - 1;                // clamp: OOB rows only feed store-guarded outputs
    const _Float16* arow = A + (size_t)row * lda;

    v8f acc0 = {}, acc1 = {};
    for (int kt = 0; kt < Kd; kt += 32) {
        v16h af = load_a_(arow + kt, half);
        const _Float16* wrow = W + (size_t)kt * F;   // B: lane&15 -> n; elements -> k (same order as A)
        v16h bf0, bf1;
#pragma unroll
        for (int e = 0; e < 8; ++e) {
            size_t k0 = (size_t)(half * 8 + e) * F, k1 = (size_t)(16 + half * 8 + e) * F;
            bf0[e] = wrow[k0 + f0 + r]; bf0[8 + e] = wrow[k1 + f0 + r];
            bf1[e] = wrow[k0 + f1 + r]; bf1[8 + e] = wrow[k1 + f1 + r];
        }
        acc0 = __builtin_amdgcn_wmma_f32_16x16x32_f16(false, af, false, bf0, (short)0, acc0, false, false);
        acc1 = __builtin_amdgcn_wmma_f32_16x16x32_f16(false, af, false, bf1, (short)0, acc1, false, false);
    }
#pragma unroll
    for (int rr = 0; rr < 8; ++rr) {
        int orow = mtile + rr + half * 8;
        if (orow < M) {
            int oc0 = f0 + r;
            if (oc0 < Fs) {
                float v = acc0[rr];
                if (bias)   v += bias[oc0];
                if (addend) v += addend[(size_t)orow * Fs + oc0];
                Out[(size_t)orow * F + oc0] = v;
            }
            int oc1 = f1 + r;
            if (oc1 < Fs) {
                float v = acc1[rr];
                if (bias)   v += bias[oc1];
                if (addend) v += addend[(size_t)orow * Fs + oc1];
                Out[(size_t)orow * F + oc1] = v;
            }
        }
    }
}

// f32 -> f16 conversion; generic version zero-pads rows and columns
__global__ void cvt_f16_kernel(const float* __restrict__ src, _Float16* __restrict__ dst, int count) {
    int i = blockIdx.x * blockDim.x + threadIdx.x;
    if (i < count) dst[i] = (_Float16)src[i];
}
__global__ void cvt_pad2_kernel(const float* __restrict__ src, _Float16* __restrict__ dst,
                                int srcR, int srcC, int dstR, int dstC) {
    int i = blockIdx.x * blockDim.x + threadIdx.x;
    if (i < dstR * dstC) {
        int rr = i / dstC, cc = i % dstC;
        dst[i] = (rr < srcR && cc < srcC) ? (_Float16)src[rr * srcC + cc] : (_Float16)0.0f;
    }
}

// ---------------------------------------------------------------------------
// K-NN: one wave per query row. Per-lane sorted top-32 insertion lists over a
// strided scan, then a stable 32-way merge on lane 0 (matches top_k ties).
// ---------------------------------------------------------------------------
__global__ void knn_kernel(const float* __restrict__ pos, const unsigned char* __restrict__ mask,
                           const int* __restrict__ batch, int* __restrict__ nbr) {
    int n = blockIdx.x;
    int lane = threadIdx.x;
    __shared__ float sd[32 * K_];
    __shared__ int   si[32 * K_];
    float* myd = sd + lane * K_;
    int*   myi = si + lane * K_;
    const float INF = __builtin_inff();

    float cx = pos[(n * A_ + 1) * 3 + 0];
    float cy = pos[(n * A_ + 1) * 3 + 1];
    float cz = pos[(n * A_ + 1) * 3 + 2];
    bool vn = mask[n] != 0;
    int  bn = batch[n];

    int cnt = 0;
    for (int j = lane; j < N_; j += 32) {
        bool ok = vn && (mask[j] != 0) && (batch[j] == bn);
        float dx = cx - pos[(j * A_ + 1) * 3 + 0];
        float dy = cy - pos[(j * A_ + 1) * 3 + 1];
        float dz = cz - pos[(j * A_ + 1) * 3 + 2];
        float d2 = ok ? (dx * dx + dy * dy + dz * dz) : INF;
        if (cnt < K_) {
            int ip = cnt++;
            while (ip > 0 && myd[ip - 1] > d2) { myd[ip] = myd[ip - 1]; myi[ip] = myi[ip - 1]; --ip; }
            myd[ip] = d2; myi[ip] = j;
        } else if (d2 < myd[K_ - 1]) {
            int ip = K_ - 1;
            while (ip > 0 && myd[ip - 1] > d2) { myd[ip] = myd[ip - 1]; myi[ip] = myi[ip - 1]; --ip; }
            myd[ip] = d2; myi[ip] = j;
        }
    }
    __syncthreads();
    if (lane == 0) {
        int head[32];
        for (int t = 0; t < 32; ++t) head[t] = 0;
        for (int kk = 0; kk < K_; ++kk) {
            float best = INF; int bi = 0x7fffffff; int bt = 0;
            for (int t = 0; t < 32; ++t) {
                if (head[t] < K_) {
                    float d = sd[t * K_ + head[t]];
                    int  ix = si[t * K_ + head[t]];
                    if (d < best || (d == best && ix < bi)) { best = d; bi = ix; bt = t; }
                }
            }
            nbr[n * K_ + kk] = bi;
            head[bt]++;
        }
    }
}

// lf = rms-normalized rel_self -> f16, padded to stride 64 (cols 60..63 = 0)
__global__ void lf_kernel(const float* __restrict__ pos, _Float16* __restrict__ lf) {
    int n = blockIdx.x * blockDim.x + threadIdx.x;
    if (n >= N_) return;
    float buf[60];
    float ss = 0.0f;
#pragma unroll
    for (int a = 0; a < A_; ++a)
#pragma unroll
        for (int c = 0; c < 3; ++c) {
            float v = pos[(n * A_ + a) * 3 + c] - pos[(n * A_ + 1) * 3 + c];
            buf[a * 3 + c] = v; ss += v * v;
        }
    float inv = rsqrtf(1e-6f + ss / 60.0f);
#pragma unroll
    for (int i = 0; i < 60; ++i) lf[n * 64 + i] = (_Float16)(buf[i] * inv);
#pragma unroll
    for (int i = 60; i < 64; ++i) lf[n * 64 + i] = (_Float16)0.0f;
}

// ---------------------------------------------------------------------------
// Fused per-node edge-feature kernel: 836 geometric features for 8 neighbours
// at a time in LDS, 836->32 projection + gathered pair row + li/lj, then
// LN + MLP(32->64->32). One block (256 threads) per node. ~43KB LDS.
// ---------------------------------------------------------------------------
#define NKC 8
__global__ void pair_kernel(const float* __restrict__ pos, const float* __restrict__ pair,
                            const int* __restrict__ nbr,
                            const float* __restrict__ li, const float* __restrict__ lj,
                            const float* __restrict__ w_dir, const float* __restrict__ w_or,
                            const float* __restrict__ w_vec, const float* __restrict__ w_dist,
                            const float* __restrict__ ln_g, const float* __restrict__ ln_b,
                            const float* __restrict__ w1, const float* __restrict__ b1,
                            const float* __restrict__ w2, const float* __restrict__ b2,
                            float* __restrict__ pout) {
    int n = blockIdx.x;
    int t = threadIdx.x;

    __shared__ float posn[60], sdn0[60];
    __shared__ float posj[NKC][60], sdj[NKC][60];
    __shared__ float dnk[NKC];
    __shared__ int   jidx[NKC];
    __shared__ float feat[NKC][836];
    __shared__ float ppre[K_][P_];
    __shared__ float hmid[K_][64];

    if (t < 60) posn[t] = pos[n * A_ * 3 + t];
    __syncthreads();
    if (t < 60) {
        int a = t / 3;
        float vx = posn[a * 3 + 0] - posn[3];
        float vy = posn[a * 3 + 1] - posn[4];
        float vz = posn[a * 3 + 2] - posn[5];
        float inv = rsqrtf(vx * vx + vy * vy + vz * vz + 1e-12f);
        sdn0[t] = (posn[t] - posn[3 + t % 3]) * inv;
    }
    __syncthreads();

    for (int kb = 0; kb < K_; kb += NKC) {
        int kk  = t >> 5;
        int t32 = t & 31;
        if (t32 == 0) jidx[kk] = nbr[n * K_ + kb + kk];
        __syncthreads();
        int j = jidx[kk];
        for (int i = t32; i < 60; i += 32) posj[kk][i] = pos[j * A_ * 3 + i];
        __syncthreads();
        for (int i = t32; i < 60; i += 32) {
            int a = i / 3;
            float vx = posj[kk][a * 3 + 0] - posj[kk][3];
            float vy = posj[kk][a * 3 + 1] - posj[kk][4];
            float vz = posj[kk][a * 3 + 2] - posj[kk][5];
            float inv = rsqrtf(vx * vx + vy * vy + vz * vz + 1e-12f);
            sdj[kk][i] = (posj[kk][i] - posj[kk][3 + i % 3]) * inv;
        }
        if (t32 == 0) {
            float dx = posn[3] - posj[kk][3];
            float dy = posn[4] - posj[kk][4];
            float dz = posn[5] - posj[kk][5];
            dnk[kk] = sqrtf(dx * dx + dy * dy + dz * dz + 1e-12f);
        }
        __syncthreads();
        for (int f = t32; f < 300; f += 32) {
            int a1 = f / 60, rem = f % 60, a2 = rem / 3, c = rem % 3;
            float v0 = posn[a1 * 3 + 0] - posj[kk][a2 * 3 + 0];
            float v1 = posn[a1 * 3 + 1] - posj[kk][a2 * 3 + 1];
            float v2 = posn[a1 * 3 + 2] - posj[kk][a2 * 3 + 2];
            float inv = rsqrtf(v0 * v0 + v1 * v1 + v2 * v2 + 1e-12f);
            float vc = (c == 0) ? v0 : ((c == 1) ? v1 : v2);
            feat[kk][f] = vc * inv;
        }
        for (int g = t32; g < 100; g += 32) {
            int a1 = g / 20, a2 = g % 20;
            float acc = 0.0f;
#pragma unroll
            for (int c = 0; c < 3; ++c) acc += sdn0[a1 * 3 + c] * sdj[kk][a2 * 3 + c];
            feat[kk][300 + g] = acc;
        }
        for (int g = t32; g < 300; g += 32) {
            int a1 = g / 60, rem = g % 60, a2 = rem / 3, c = rem % 3;
            float u0 = sdn0[a1 * 3 + 0], u1 = sdn0[a1 * 3 + 1], u2 = sdn0[a1 * 3 + 2];
            float v0 = sdj[kk][a2 * 3 + 0], v1 = sdj[kk][a2 * 3 + 1], v2 = sdj[kk][a2 * 3 + 2];
            float rv = (c == 0) ? (u1 * v2 - u2 * v1)
                     : (c == 1) ? (u2 * v0 - u0 * v2)
                                : (u0 * v1 - u1 * v0);
            feat[kk][400 + g] = rv;
        }
        for (int g = t32; g < 120; g += 32) {
            int a = g / 6, cc = g % 6;
            float v;
            if (cc < 3) v = 0.1f * (posn[a * 3 + cc] - posn[3 + cc]);
            else { int c2 = cc - 3; v = 0.1f * (posj[kk][a * 3 + c2] - posn[3 + c2]); }
            feat[kk][700 + g] = v;
        }
        if (t32 < NBINS_) {
            float center = 22.0f * (float)t32 / 15.0f;
            float z = (dnk[kk] - center) / 1.375f;
            feat[kk][820 + t32] = expf(-0.5f * z * z);
        }
        __syncthreads();
        {
            int o = t & 31, kl = t >> 5, kg = kb + kl, jn = jidx[kl];
            float acc = pair[((size_t)n * N_ + jn) * P_ + o] + li[n * P_ + o] + lj[jn * P_ + o];
            for (int i = 0; i < 300; ++i) acc += feat[kl][i]       * w_dir[i * P_ + o];
            for (int i = 0; i < 400; ++i) acc += feat[kl][300 + i] * w_or [i * P_ + o];
            for (int i = 0; i < 120; ++i) acc += feat[kl][700 + i] * w_vec[i * P_ + o];
            for (int i = 0; i < 16;  ++i) acc += feat[kl][820 + i] * w_dist[i * P_ + o];
            ppre[kg][o] = acc;
        }
        __syncthreads();
    }
    if (t < K_) {
        float m = 0.0f;
        for (int o = 0; o < P_; ++o) m += ppre[t][o];
        m /= P_;
        float v = 0.0f;
        for (int o = 0; o < P_; ++o) { float d = ppre[t][o] - m; v += d * d; }
        v /= P_;
        float inv = rsqrtf(v + 1e-5f);
        for (int o = 0; o < P_; ++o) ppre[t][o] = (ppre[t][o] - m) * inv * ln_g[o] + ln_b[o];
    }
    __syncthreads();
    for (int idx = t; idx < K_ * 64; idx += 256) {
        int k = idx >> 6, o = idx & 63;
        float acc = b1[o];
        for (int i = 0; i < P_; ++i) acc += ppre[k][i] * w1[i * 64 + o];
        hmid[k][o] = geluf_(acc);
    }
    __syncthreads();
    for (int idx = t; idx < K_ * P_; idx += 256) {
        int k = idx >> 5, o = idx & 31;
        float acc = b2[o];
        for (int i = 0; i < 64; ++i) acc += hmid[k][i] * w2[i * P_ + o];
        pout[((size_t)n * K_ + k) * P_ + o] = acc;
    }
}

// cond_norm: x = sigmoid(s) * LN(local) + t  -> f16 (feeds GEMMs only)
__global__ void cond_norm_kernel(const float* __restrict__ xin, const float* __restrict__ s,
                                 const float* __restrict__ tt, _Float16* __restrict__ out) {
    int n = blockIdx.x, tid = threadIdx.x;
    __shared__ float red[C_];
    float v = xin[(size_t)n * C_ + tid];
    red[tid] = v; __syncthreads();
    for (int off = C_ / 2; off > 0; off >>= 1) {
        if (tid < off) red[tid] += red[tid + off];
        __syncthreads();
    }
    float m = red[0] / C_;
    __syncthreads();
    float d = v - m;
    red[tid] = d * d; __syncthreads();
    for (int off = C_ / 2; off > 0; off >>= 1) {
        if (tid < off) red[tid] += red[tid + off];
        __syncthreads();
    }
    float var = red[0] / C_;
    float y = d * rsqrtf(var + 1e-5f);
    size_t i = (size_t)n * C_ + tid;
    out[i] = (_Float16)(sigmoidf_(s[i]) * y + tt[i]);
}

// ---------------------------------------------------------------------------
// Attention (one block per node). op = (att^T p) @ w_pv reordering.
// cat written in f16 with stride 544 (cols 536..543 zero) to feed the wo GEMM.
// ---------------------------------------------------------------------------
__global__ void attn_kernel(const float* __restrict__ qf, const float* __restrict__ kf,
                            const float* __restrict__ vf, const float* __restrict__ p,
                            const int* __restrict__ nbr, const float* __restrict__ w_pb,
                            const float* __restrict__ w_pv, const float* __restrict__ pos,
                            const unsigned char* __restrict__ mask, _Float16* __restrict__ cat) {
    int n = blockIdx.x, t = threadIdx.x;
    __shared__ float att[H_][K_];
    __shared__ float ps[K_][P_];
    __shared__ float sph[P_][H_];
    __shared__ float rel[K_][3];
    __shared__ int   jj[K_];
    __shared__ float mv[K_];

    if (t < K_) {
        int j = nbr[n * K_ + t];
        jj[t] = j;
#pragma unroll
        for (int c = 0; c < 3; ++c)
            rel[t][c] = 0.1f * (pos[(j * A_ + 1) * 3 + c] - pos[(n * A_ + 1) * 3 + c]);
        mv[t] = (mask[n] && mask[j]) ? 1.0f : 0.0f;
    }
    for (int i = t; i < K_ * P_; i += 256)
        ps[i / P_][i % P_] = p[((size_t)n * K_ + i / P_) * P_ + (i % P_)];
    __syncthreads();
    {   // logits
        int h = t >> 5, k = t & 31, j = jj[k];
        float acc = 0.0f;
        for (int d = 0; d < DH_; ++d)
            acc += qf[(size_t)n * C_ + h * DH_ + d] * kf[(size_t)j * C_ + h * DH_ + d];
        acc *= 0.17677669529663687f;  // 1/sqrt(DH)
        for (int pp = 0; pp < P_; ++pp) acc += ps[k][pp] * w_pb[pp * H_ + h];
        att[h][k] = (mv[k] > 0.0f) ? acc : -1e9f;
    }
    __syncthreads();
    if (t < H_) {   // softmax over k
        float m = -__builtin_inff();
        for (int k = 0; k < K_; ++k) m = fmaxf(m, att[t][k]);
        float tmp[K_]; float ssum = 0.0f;
        for (int k = 0; k < K_; ++k) { float e = expf(att[t][k] - m); tmp[k] = e; ssum += e; }
        for (int k = 0; k < K_; ++k) att[t][k] = tmp[k] / ssum;
    }
    __syncthreads();
    {   // o = att @ v_gathered
        int h = t >> 5, d = t & 31;
        float acc = 0.0f;
        for (int k = 0; k < K_; ++k) acc += att[h][k] * vf[(size_t)jj[k] * C_ + h * DH_ + d];
        cat[(size_t)n * 544 + h * DH_ + d] = (_Float16)acc;
    }
    {   // sph[p][h] = sum_k att[h][k] * p[k][p]
        int pp = t >> 3, h = t & 7;
        float acc = 0.0f;
        for (int k = 0; k < K_; ++k) acc += att[h][k] * ps[k][pp];
        sph[pp][h] = acc;
    }
    __syncthreads();
    {   // op = sph^T @ w_pv slice
        int h = t >> 5, d = t & 31;
        float acc = 0.0f;
        for (int pp = 0; pp < P_; ++pp) acc += sph[pp][h] * w_pv[pp * (H_ * DH_) + h * DH_ + d];
        cat[(size_t)n * 544 + 256 + h * DH_ + d] = (_Float16)acc;
    }
    if (t < H_ * 3) {   // opos
        int h = t / 3, c = t % 3;
        float acc = 0.0f;
        for (int k = 0; k < K_; ++k) acc += att[h][k] * rel[k][c];
        cat[(size_t)n * 544 + 512 + t] = (_Float16)acc;
    }
    if (t >= 24 && t < 32)  // zero-pad cols 536..543
        cat[(size_t)n * 544 + 512 + t] = (_Float16)0.0f;
}

__global__ void gelu_gate_kernel(const float* __restrict__ g, const float* __restrict__ v,
                                 _Float16* __restrict__ h, int count) {
    int i = blockIdx.x * blockDim.x + threadIdx.x;
    if (i < count) h[i] = (_Float16)(geluf_(g[i]) * v[i]);
}

__global__ void copy_kernel(const float* __restrict__ src, float* __restrict__ dst, int count) {
    int i = blockIdx.x * blockDim.x + threadIdx.x;
    if (i < count) dst[i] = src[i];
}

// pos_out: sg has stride 32 (col 0 valid), upd has stride 64 (cols 0..59 valid)
__global__ void pos_out_kernel(const float* __restrict__ pos, const float* __restrict__ sg,
                               const float* __restrict__ upd, float* __restrict__ out) {
    int n = blockIdx.x, t = threadIdx.x;
    if (t >= 60) return;
    float sc = 10.0f * sigmoidf_(sg[(size_t)n * 32]);   // SIGMA_DATA * sigmoid
    out[(size_t)n * 60 + t] = pos[(size_t)n * 60 + t] + sc * upd[(size_t)n * 64 + t];
}

// ---------------------------------------------------------------------------
static inline void gemm(const _Float16* A, int lda, const _Float16* W,
                        const float* bias, const float* addend,
                        float* Out, int M, int Kd, int F, int Fs, hipStream_t stream) {
    dim3 grid((M + 15) / 16, (F + 127) / 128);
    gemm_wmma_kernel<<<grid, 128, 0, stream>>>(A, lda, W, bias, addend, Out, M, Kd, F, Fs);
}
static inline void cvt(const float* src, _Float16* dst, int count, hipStream_t stream) {
    cvt_f16_kernel<<<(count + 255) / 256, 256, 0, stream>>>(src, dst, count);
}
static inline void cvt_pad2(const float* src, _Float16* dst, int srcR, int srcC, int dstR, int dstC,
                            hipStream_t stream) {
    int cnt = dstR * dstC;
    cvt_pad2_kernel<<<(cnt + 255) / 256, 256, 0, stream>>>(src, dst, srcR, srcC, dstR, dstC);
}

extern "C" void kernel_launch(void* const* d_in, const int* in_sizes, int n_in,
                              void* d_out, int out_size, void* d_ws, size_t ws_size,
                              hipStream_t stream) {
    (void)in_sizes; (void)n_in; (void)out_size; (void)ws_size;
    // ---- inputs (setup_inputs dict order; params expanded in insertion order)
    const float* local = (const float*)d_in[0];
    const float* pos   = (const float*)d_in[1];
    const float* pair  = (const float*)d_in[2];
    const float* cond  = (const float*)d_in[3];
    const int*           batch = (const int*)d_in[6];
    const unsigned char* mask  = (const unsigned char*)d_in[7];
    int pi = 8;
    const float* w_li   = (const float*)d_in[pi++];
    const float* w_lj   = (const float*)d_in[pi++];
    const float* w_dist = (const float*)d_in[pi++];
    const float* w_dir  = (const float*)d_in[pi++];
    const float* w_or   = (const float*)d_in[pi++];
    const float* w_vec  = (const float*)d_in[pi++];
    const float* ln_g   = (const float*)d_in[pi++];
    const float* ln_b   = (const float*)d_in[pi++];
    const float* mlp_w1 = (const float*)d_in[pi++];
    const float* mlp_b1 = (const float*)d_in[pi++];
    const float* mlp_w2 = (const float*)d_in[pi++];
    const float* mlp_b2 = (const float*)d_in[pi++];
    const float* w_lf   = (const float*)d_in[pi++];
    const float* b_lf   = (const float*)d_in[pi++];
    const float* cn1ws = (const float*)d_in[pi++]; const float* cn1bs = (const float*)d_in[pi++];
    const float* cn1wb = (const float*)d_in[pi++]; const float* cn1bb = (const float*)d_in[pi++];
    const float* cn2ws = (const float*)d_in[pi++]; const float* cn2bs = (const float*)d_in[pi++];
    const float* cn2wb = (const float*)d_in[pi++]; const float* cn2bb = (const float*)d_in[pi++];
    const float* cn3ws = (const float*)d_in[pi++]; const float* cn3bs = (const float*)d_in[pi++];
    const float* cn3wb = (const float*)d_in[pi++]; const float* cn3bb = (const float*)d_in[pi++];
    const float* wq    = (const float*)d_in[pi++];
    const float* wk    = (const float*)d_in[pi++];
    const float* wv    = (const float*)d_in[pi++];
    const float* w_pb  = (const float*)d_in[pi++];
    const float* w_pv  = (const float*)d_in[pi++];
    const float* wo    = (const float*)d_in[pi++];
    const float* bo    = (const float*)d_in[pi++];
    const float* m2wg  = (const float*)d_in[pi++];
    const float* m2bg  = (const float*)d_in[pi++];
    const float* m2wv  = (const float*)d_in[pi++];
    const float* m2bv  = (const float*)d_in[pi++];
    const float* m2wo  = (const float*)d_in[pi++];
    const float* m2bo  = (const float*)d_in[pi++];
    const float* w_scale = (const float*)d_in[pi++];
    const float* b_scale = (const float*)d_in[pi++];
    const float* w_pos   = (const float*)d_in[pi++];
    const float* b_pos   = (const float*)d_in[pi++];

    // ---- workspace layout
    float* ws = (float*)d_ws;
    size_t off = 0;
    auto allocF = [&](size_t cnt) { float* p = ws + off; off += cnt; return p; };
    auto allocH = [&](size_t cnt) { _Float16* p = (_Float16*)(ws + off); off += (cnt + 1) / 2; return p; };

    int*   nbr    = (int*)allocF((size_t)N_ * K_);
    float* li     = allocF((size_t)N_ * P_);
    float* ljb    = allocF((size_t)N_ * P_);
    float* pbuf   = allocF((size_t)N_ * K_ * P_);
    float* local1 = allocF((size_t)N_ * C_);
    float* sbuf   = allocF((size_t)N_ * C_);
    float* tbuf   = allocF((size_t)N_ * C_);
    float* qfb    = allocF((size_t)N_ * C_);
    float* kfb    = allocF((size_t)N_ * C_);
    float* vfb    = allocF((size_t)N_ * C_);
    float* local2 = allocF((size_t)N_ * C_);
    float* gbuf   = allocF((size_t)N_ * 2 * C_);
    float* v2buf  = allocF((size_t)N_ * 2 * C_);
    float* local3 = allocF((size_t)N_ * C_);
    float* sgb    = allocF((size_t)N_ * 32);      // F padded to 32 (col 0 real)
    float* updb   = allocF((size_t)N_ * 64);      // F padded to 64 (cols 0..59 real)
    // f16 activations (GEMM A operands)
    _Float16* local_h = allocH((size_t)N_ * C_);
    _Float16* cond_h  = allocH((size_t)N_ * C_);
    _Float16* lfb_h   = allocH((size_t)N_ * 64);
    _Float16* xbuf_h  = allocH((size_t)N_ * C_);
    _Float16* catb_h  = allocH((size_t)N_ * 544);
    _Float16* ybuf_h  = allocH((size_t)N_ * C_);
    _Float16* h2buf_h = allocH((size_t)N_ * 2 * C_);
    _Float16* lnout_h = allocH((size_t)N_ * C_);
    // f16 weights (GEMM B operands), padded so Kd % 32 == 0 and F % 32 == 0
    _Float16* w_li_h  = allocH(C_ * P_);
    _Float16* w_lj_h  = allocH(C_ * P_);
    _Float16* w_lf_h  = allocH(64 * C_);          // rows 60 -> 64
    _Float16* cn1ws_h = allocH(C_ * C_);
    _Float16* cn1wb_h = allocH(C_ * C_);
    _Float16* cn2ws_h = allocH(C_ * C_);
    _Float16* cn2wb_h = allocH(C_ * C_);
    _Float16* cn3ws_h = allocH(C_ * C_);
    _Float16* cn3wb_h = allocH(C_ * C_);
    _Float16* wq_h    = allocH(C_ * C_);
    _Float16* wk_h    = allocH(C_ * C_);
    _Float16* wv_h    = allocH(C_ * C_);
    _Float16* wo_h    = allocH(544 * C_);         // rows 536 -> 544
    _Float16* m2wg_h  = allocH(C_ * 2 * C_);
    _Float16* m2wv_h  = allocH(C_ * 2 * C_);
    _Float16* m2wo_h  = allocH(2 * C_ * C_);
    _Float16* w_scale_h = allocH(C_ * 32);        // cols 1 -> 32
    _Float16* w_pos_h   = allocH(C_ * 64);        // cols 60 -> 64

    // ---- one-time (per launch) f16 conversions
    cvt(local, local_h, N_ * C_, stream);
    cvt(cond,  cond_h,  N_ * C_, stream);
    cvt(w_li, w_li_h, C_ * P_, stream);
    cvt(w_lj, w_lj_h, C_ * P_, stream);
    cvt_pad2(w_lf, w_lf_h, 60, C_, 64, C_, stream);
    cvt(cn1ws, cn1ws_h, C_ * C_, stream);  cvt(cn1wb, cn1wb_h, C_ * C_, stream);
    cvt(cn2ws, cn2ws_h, C_ * C_, stream);  cvt(cn2wb, cn2wb_h, C_ * C_, stream);
    cvt(cn3ws, cn3ws_h, C_ * C_, stream);  cvt(cn3wb, cn3wb_h, C_ * C_, stream);
    cvt(wq, wq_h, C_ * C_, stream);
    cvt(wk, wk_h, C_ * C_, stream);
    cvt(wv, wv_h, C_ * C_, stream);
    cvt_pad2(wo, wo_h, 536, C_, 544, C_, stream);
    cvt(m2wg, m2wg_h, C_ * 2 * C_, stream);
    cvt(m2wv, m2wv_h, C_ * 2 * C_, stream);
    cvt(m2wo, m2wo_h, 2 * C_ * C_, stream);
    cvt_pad2(w_scale, w_scale_h, C_, 1,  C_, 32, stream);
    cvt_pad2(w_pos,   w_pos_h,   C_, 60, C_, 64, stream);

    // ---- pipeline (all stream-ordered)
    knn_kernel<<<N_, 32, 0, stream>>>(pos, mask, batch, nbr);
    gemm(local_h, C_, w_li_h, nullptr, nullptr, li,  N_, C_, P_, P_, stream);
    gemm(local_h, C_, w_lj_h, nullptr, nullptr, ljb, N_, C_, P_, P_, stream);
    lf_kernel<<<(N_ + 127) / 128, 128, 0, stream>>>(pos, lfb_h);
    gemm(lfb_h, 64, w_lf_h, b_lf, local, local1, N_, 64, C_, C_, stream);  // local1 = local + lf@w_lf + b_lf
    pair_kernel<<<N_, 256, 0, stream>>>(pos, pair, nbr, li, ljb, w_dir, w_or, w_vec,
                                        w_dist, ln_g, ln_b, mlp_w1, mlp_b1, mlp_w2, mlp_b2, pbuf);
    // cond_norm 1 -> x ; q,k,v
    gemm(cond_h, C_, cn1ws_h, cn1bs, nullptr, sbuf, N_, C_, C_, C_, stream);
    gemm(cond_h, C_, cn1wb_h, cn1bb, nullptr, tbuf, N_, C_, C_, C_, stream);
    cond_norm_kernel<<<N_, C_, 0, stream>>>(local1, sbuf, tbuf, xbuf_h);
    gemm(xbuf_h, C_, wq_h, nullptr, nullptr, qfb, N_, C_, C_, C_, stream);
    gemm(xbuf_h, C_, wk_h, nullptr, nullptr, kfb, N_, C_, C_, C_, stream);
    gemm(xbuf_h, C_, wv_h, nullptr, nullptr, vfb, N_, C_, C_, C_, stream);
    attn_kernel<<<N_, 256, 0, stream>>>(qfb, kfb, vfb, pbuf, nbr, w_pb, w_pv, pos, mask, catb_h);
    gemm(catb_h, 544, wo_h, bo, local1, local2, N_, 544, C_, C_, stream);  // local2 = local1 + cat@wo + bo
    // cond_norm 2 -> y ; gated MLP
    gemm(cond_h, C_, cn2ws_h, cn2bs, nullptr, sbuf, N_, C_, C_, C_, stream);
    gemm(cond_h, C_, cn2wb_h, cn2bb, nullptr, tbuf, N_, C_, C_, C_, stream);
    cond_norm_kernel<<<N_, C_, 0, stream>>>(local2, sbuf, tbuf, ybuf_h);
    gemm(ybuf_h, C_, m2wg_h, m2bg, nullptr, gbuf,  N_, C_, 2 * C_, 2 * C_, stream);
    gemm(ybuf_h, C_, m2wv_h, m2bv, nullptr, v2buf, N_, C_, 2 * C_, 2 * C_, stream);
    gelu_gate_kernel<<<(N_ * 2 * C_ + 255) / 256, 256, 0, stream>>>(gbuf, v2buf, h2buf_h, N_ * 2 * C_);
    gemm(h2buf_h, 2 * C_, m2wo_h, m2bo, local2, local3, N_, 2 * C_, C_, C_, stream);
    // cond_norm 3 -> heads
    gemm(cond_h, C_, cn3ws_h, cn3bs, nullptr, sbuf, N_, C_, C_, C_, stream);
    gemm(cond_h, C_, cn3wb_h, cn3bb, nullptr, tbuf, N_, C_, C_, C_, stream);
    cond_norm_kernel<<<N_, C_, 0, stream>>>(local3, sbuf, tbuf, lnout_h);
    gemm(lnout_h, C_, w_scale_h, b_scale, nullptr, sgb,  N_, C_, 32, 1,  stream);
    gemm(lnout_h, C_, w_pos_h,   b_pos,   nullptr, updb, N_, C_, 64, 60, stream);
    // outputs: [local (N*C) | pos (N*A*3)] f32
    copy_kernel<<<(N_ * C_ + 255) / 256, 256, 0, stream>>>(local3, (float*)d_out, N_ * C_);
    pos_out_kernel<<<N_, 64, 0, stream>>>(pos, sgb, updb, (float*)d_out + (size_t)N_ * C_);
}
